// TimeMixer_6347961664093
// MI455X (gfx1250) — matrix operand order repaired and checked by reading, not compiled, not run on hardware
//
#include <hip/hip_runtime.h>

typedef __attribute__((ext_vector_type(16))) _Float16 v16h;
typedef __attribute__((ext_vector_type(8)))  _Float16 v8h;
typedef __attribute__((ext_vector_type(2)))  _Float16 v2h;
typedef __attribute__((ext_vector_type(8)))  float    v8f;
typedef __attribute__((ext_vector_type(4)))  float    v4f;

#define T_DIM 128
#define F_DIM 256
#define XT_STRIDE 136   // halves per row of Xt/Hs (128 + 8 pad -> bank spread)
#define W_STRIDE  136   // halves per row of weight tiles

// LDS layout (bytes)
#define OFF_XT   0
#define OFF_HS   69632      // 256*136*2
#define OFF_L1   139264     // + 256*136*2
#define OFF_L2   174080     // + 128*136*2
#define OFF_B1   208896     // + 128*136*2
#define OFF_B2   209408
#define OFF_RED  209920
#define SMEM_BYTES 210048   // 209920 + 16*4 + pad

union AFill { v16h v; v8h h[2]; };

__global__ __launch_bounds__(256)
void timemixer_kernel(const float* __restrict__ x,
                      const float* __restrict__ ln_w,
                      const float* __restrict__ ln_b,
                      const float* __restrict__ W1,
                      const float* __restrict__ b1v,
                      const float* __restrict__ W2,
                      const float* __restrict__ b2v,
                      float* __restrict__ out)
{
    extern __shared__ char smem[];
    _Float16* Xt = (_Float16*)(smem + OFF_XT);   // [F][XT_STRIDE] normalized, transposed
    _Float16* Hs = (_Float16*)(smem + OFF_HS);   // [F][XT_STRIDE] hardswish(h1)
    _Float16* L1 = (_Float16*)(smem + OFF_L1);   // [T][W_STRIDE]  tril(W1), row s, t contig
    _Float16* L2 = (_Float16*)(smem + OFF_L2);   // [T][W_STRIDE]  tril(W2)
    float*   sb1 = (float*)(smem + OFF_B1);
    float*   sb2 = (float*)(smem + OFF_B2);
    float*   red = (float*)(smem + OFF_RED);     // 16 floats (per-wave partials)

    const int tid  = threadIdx.x;
    const int bc   = blockIdx.x;                 // 0..511 = b*C+c
    const int wv   = tid >> 5;                   // wave 0..7
    const int lane = tid & 31;
    const int lrow = lane & 15;
    const int hi   = lane >> 4;                  // 0 or 1 (lane half)

    const float* xs = x   + (size_t)bc * (T_DIM * F_DIM);
    float*       ys = out + (size_t)bc * (T_DIM * F_DIM);

    // ---- weights -> LDS, tril-masked, f16 (2 elems/thread/iter) ---------
    #pragma unroll 4
    for (int i = 0; i < 32; ++i) {
        int e = (i * 256 + tid) * 2;             // 16384 elements, pairs share a row
        int s = e >> 7;
        int t = e & 127;
        v2h p1, p2;
        p1.x = (_Float16)((t     <= s) ? W1[e]     : 0.0f);
        p1.y = (_Float16)((t + 1 <= s) ? W1[e + 1] : 0.0f);
        p2.x = (_Float16)((t     <= s) ? W2[e]     : 0.0f);
        p2.y = (_Float16)((t + 1 <= s) ? W2[e + 1] : 0.0f);
        *(v2h*)(L1 + s * W_STRIDE + t) = p1;
        *(v2h*)(L2 + s * W_STRIDE + t) = p2;
    }
    if (tid < 128) { sb1[tid] = b1v[tid]; sb2[tid] = b2v[tid]; }

    // ---- LayerNorm pass 1: mean / var over T*F = 32768 ------------------
    float sum = 0.f, sq = 0.f;
    #pragma unroll 4
    for (int i = 0; i < 32; ++i) {
        v4f v = *(const v4f*)(xs + (i * 256 + tid) * 4);
        sum += v.x + v.y + v.z + v.w;
        sq  += v.x * v.x + v.y * v.y + v.z * v.z + v.w * v.w;
    }
    #pragma unroll
    for (int o = 16; o > 0; o >>= 1) {           // wave32 butterfly reduce
        sum += __shfl_xor(sum, o, 32);
        sq  += __shfl_xor(sq,  o, 32);
    }
    if (lane == 0) { red[wv] = sum; red[8 + wv] = sq; }
    __syncthreads();
    float tot = 0.f, totq = 0.f;
    #pragma unroll
    for (int i = 0; i < 8; ++i) { tot += red[i]; totq += red[8 + i]; }
    const float mu  = tot * (1.0f / 32768.0f);
    const float var = totq * (1.0f / 32768.0f) - mu * mu;
    const float rs  = rsqrtf(var + 1e-5f);

    // ---- pass 2: normalize + transpose into Xt (f16, 2 t per store) -----
    #pragma unroll 4
    for (int t = 0; t < T_DIM; t += 2) {
        int i0 = t * F_DIM + tid;                // f = tid, coalesced
        int i1 = i0 + F_DIM;
        v2h p;
        p.x = (_Float16)((xs[i0] - mu) * rs * ln_w[i0] + ln_b[i0]);
        p.y = (_Float16)((xs[i1] - mu) * rs * ln_w[i1] + ln_b[i1]);
        *(v2h*)(Xt + tid * XT_STRIDE + t) = p;
    }
    __syncthreads();

    // ---- fused triangular matmuls via WMMA ------------------------------
    // Wave owns f-strips mt = wv and wv+8 for BOTH matmuls (no barrier needed:
    // per-wave LDS ordering covers the Hs write->read dependency).
    const int fA0 = wv * 16 + lrow;              // A row, strip 0
    const int fA1 = fA0 + 128;                   // A row, strip 1 (= (wv+8)*16+lrow)

    // ======== matmul1: h = Xt @ tril(W1)^T + b1, hardswish -> Hs ========
    {
        AFill a[2][4];
        #pragma unroll
        for (int k = 0; k < 4; ++k) {
            const _Float16* p0 = Xt + fA0 * XT_STRIDE + k * 32 + hi * 8;
            a[0][k].h[0] = *(const v8h*)p0;        // K = k*32 + hi*8 + 0..7
            a[0][k].h[1] = *(const v8h*)(p0 + 16); // K = k*32 + 16 + hi*8 + 0..7
            const _Float16* p1 = Xt + fA1 * XT_STRIDE + k * 32 + hi * 8;
            a[1][k].h[0] = *(const v8h*)p1;
            a[1][k].h[1] = *(const v8h*)(p1 + 16);
        }
        #pragma unroll
        for (int nt = 0; nt < 8; ++nt) {
            const int kmax = nt >> 1;            // tril: B chunk zero for k > nt/2
            const int srow = nt * 16 + lrow;     // B column N = s
            AFill b[4];
            #pragma unroll
            for (int k = 0; k <= kmax; ++k) {
                const _Float16* p = L1 + srow * W_STRIDE + k * 32 + hi * 8;
                b[k].h[0] = *(const v8h*)p;       // K = k*32 + hi*8 + 0..7
                b[k].h[1] = *(const v8h*)(p + 16); // K = k*32 + 16 + hi*8 + 0..7
            }
            const int tcol = nt * 16 + lrow;     // output t index (N lane)
            const float bb = sb1[tcol];
            #pragma unroll
            for (int mi = 0; mi < 2; ++mi) {
                v8f acc = {};
                #pragma unroll
                for (int k = 0; k <= kmax; ++k)
                    acc = __builtin_amdgcn_wmma_f32_16x16x32_f16(
                        false, a[mi][k].v, false, b[k].v, (short)0, acc, false, false);
                const int fbase = (wv + mi * 8) * 16 + hi * 8;
                #pragma unroll
                for (int r = 0; r < 8; ++r) {    // D row M = r + hi*8
                    float v = acc[r] + bb;
                    float g = fminf(fmaxf(v + 3.0f, 0.0f), 6.0f);
                    v = v * g * (1.0f / 6.0f);   // hardswish
                    Hs[(fbase + r) * XT_STRIDE + tcol] = (_Float16)v;
                }
            }
        }
    }

    // ======== matmul2: y = Xt + Hs @ tril(W2)^T + b2 -> global ==========
    {
        AFill a[2][4];
        #pragma unroll
        for (int k = 0; k < 4; ++k) {
            const _Float16* p0 = Hs + fA0 * XT_STRIDE + k * 32 + hi * 8;
            a[0][k].h[0] = *(const v8h*)p0;
            a[0][k].h[1] = *(const v8h*)(p0 + 16);
            const _Float16* p1 = Hs + fA1 * XT_STRIDE + k * 32 + hi * 8;
            a[1][k].h[0] = *(const v8h*)p1;
            a[1][k].h[1] = *(const v8h*)(p1 + 16);
        }
        #pragma unroll
        for (int nt = 0; nt < 8; ++nt) {
            const int kmax = nt >> 1;
            const int srow = nt * 16 + lrow;
            AFill b[4];
            #pragma unroll
            for (int k = 0; k <= kmax; ++k) {
                const _Float16* p = L2 + srow * W_STRIDE + k * 32 + hi * 8;
                b[k].h[0] = *(const v8h*)p;
                b[k].h[1] = *(const v8h*)(p + 16);
            }
            const int tcol = nt * 16 + lrow;
            const float bb = sb2[tcol];
            #pragma unroll
            for (int mi = 0; mi < 2; ++mi) {
                v8f acc = {};
                #pragma unroll
                for (int k = 0; k <= kmax; ++k)
                    acc = __builtin_amdgcn_wmma_f32_16x16x32_f16(
                        false, a[mi][k].v, false, b[k].v, (short)0, acc, false, false);
                const int fbase = (wv + mi * 8) * 16 + hi * 8;
                float o[8];
                #pragma unroll
                for (int r = 0; r < 8; ++r)
                    o[r] = (float)Xt[(fbase + r) * XT_STRIDE + tcol] + acc[r] + bb;
                // out[t][f]: lane has fixed t=tcol, 8 contiguous f values
                float* po = ys + tcol * F_DIM + fbase;
                *(v4f*)po       = (v4f){o[0], o[1], o[2], o[3]};
                *(v4f*)(po + 4) = (v4f){o[4], o[5], o[6], o[7]};
            }
        }
    }
}

extern "C" void kernel_launch(void* const* d_in, const int* in_sizes, int n_in,
                              void* d_out, int out_size, void* d_ws, size_t ws_size,
                              hipStream_t stream) {
    (void)in_sizes; (void)n_in; (void)out_size; (void)d_ws; (void)ws_size;
    const float* x    = (const float*)d_in[0];
    const float* ln_w = (const float*)d_in[1];
    const float* ln_b = (const float*)d_in[2];
    const float* W1   = (const float*)d_in[3];
    const float* b1   = (const float*)d_in[4];
    const float* W2   = (const float*)d_in[5];
    const float* b2   = (const float*)d_in[6];
    float* out = (float*)d_out;

    hipFuncSetAttribute((const void*)timemixer_kernel,
                        hipFuncAttributeMaxDynamicSharedMemorySize, SMEM_BYTES);
    // one workgroup per (b,c) slab: 64*8 = 512
    timemixer_kernel<<<512, 256, SMEM_BYTES, stream>>>(x, ln_w, ln_b, W1, b1, W2, b2, out);
}
